// LSTMEncoder_16569983828036
// MI455X (gfx1250) — matrix operand order repaired and checked by reading, not compiled, not run on hardware
//
#include <hip/hip_runtime.h>
#include <hip/hip_bf16.h>

typedef __attribute__((ext_vector_type(16))) _Float16 v16h;
typedef __attribute__((ext_vector_type(8)))  _Float16 v8h;
typedef __attribute__((ext_vector_type(4)))  _Float16 v4h;
typedef __attribute__((ext_vector_type(8)))  float    v8f;

#define T_LEN 512
#define B_SZ  512
#define D_IN  35
#define H1    64
#define G1    256   // 4*H1
#define H2    3
#define G2    12    // 4*H2
#define K2    128   // 2*H1

// workspace layout (bytes)
#define XP1_BYTES  (2ull*T_LEN*32*16*256*2)          // 268435456  (f16 fragments)
#define OUT1_BYTES ((size_t)B_SZ*T_LEN*128*2)        // 67108864   (f16)
#define XP2_BYTES  (2ull*T_LEN*B_SZ*12*4)            // 25165824   (f32)
#define OUT1_OFF   XP1_BYTES
#define XP2_OFF    (XP1_BYTES + OUT1_BYTES)

__device__ __forceinline__ v8f wmma16(v16h a, v16h b, v8f c) {
    return __builtin_amdgcn_wmma_f32_16x16x32_f16(false, a, false, b, (short)0, c, false, false);
}
__device__ __forceinline__ float sigf(float x) {
    return __fdividef(1.0f, 1.0f + __expf(-x));
}
__device__ __forceinline__ float tanh_fast(float x) {
    return __fdividef(2.0f, 1.0f + __expf(-2.0f * x)) - 1.0f;
}

// xp1 fragment address (in halves): [dir][t][btile][ntile][lane][8]
__device__ __forceinline__ size_t xp1_base(int dir, int t, int bt, int nt, int lane) {
    return ((((size_t)dir * T_LEN + t) * 32 + bt) * 16 + nt) * 256 + (size_t)lane * 8;
}

// ---------------------------------------------------------------------------
// Kernel 1: layer-1 input projection. xp1 = x @ W_ih^T + b_ih + b_hh,
// stored as f16 WMMA C/D fragments, M-tiles = 16 batch rows at fixed t.
// grid (32 tchunks, 32 btiles, 2 dirs), 256 threads (8 waves, 2 ntiles each)
// ---------------------------------------------------------------------------
__global__ __launch_bounds__(256)
void lstm1_proj(const float* __restrict__ x,
                const float* __restrict__ w_ih_f,
                const float* __restrict__ b_ih_f, const float* __restrict__ b_hh_f,
                const float* __restrict__ w_ih_b,
                const float* __restrict__ b_ih_b, const float* __restrict__ b_hh_b,
                _Float16* __restrict__ xp1)
{
    const int dir = blockIdx.z, bt = blockIdx.y, tch = blockIdx.x;
    const float* w  = dir ? w_ih_b : w_ih_f;
    const float* bi = dir ? b_ih_b : b_ih_f;
    const float* bh = dir ? b_hh_b : b_hh_f;
    const int lane = threadIdx.x & 31, wv = threadIdx.x >> 5;
    const int nif = lane & 15, khi = lane >> 4;

    // B fragments from W_ih (B[k][n] = w[g][k]); K padded 35 -> 64 (2 chunks)
    v16h Bf[2][2];
    float biasv[2];
    #pragma unroll
    for (int j = 0; j < 2; j++) {
        int g = (wv * 2 + j) * 16 + nif;
        biasv[j] = bi[g] + bh[g];
        {   // chunk 0: k = khi*8 + e + (e&8), always < 32 < 35
            int kb = khi * 8;
            v16h f;
            #pragma unroll
            for (int e = 0; e < 16; e++) f[e] = (_Float16)w[g * D_IN + kb + e + (e & 8)];
            Bf[j][0] = f;
        }
        {   // chunk 1: k = 32 + khi*8 + e + (e&8), guarded
            int kb = 32 + khi * 8;
            v16h f;
            #pragma unroll
            for (int e = 0; e < 16; e++) {
                int k = kb + e + (e & 8);
                f[e] = (k < D_IN) ? (_Float16)w[g * D_IN + k] : (_Float16)0.0f;
            }
            Bf[j][1] = f;
        }
    }

    const int row = bt * 16 + nif;     // batch row for this lane's A entries
    for (int i = 0; i < 16; i++) {
        int t = tch * 16 + i;
        const float* xr = x + ((size_t)row * T_LEN + t) * D_IN;
        // A fragments: per lane two 8-long contiguous K runs per chunk
        v16h A0, A1;
        {
            int off = khi * 8;   // chunk 0: runs [off, off+8), [off+16, off+24), all < 32
            #pragma unroll
            for (int e = 0; e < 8; e++) {
                A0[e]     = (_Float16)xr[off + e];
                A0[e + 8] = (_Float16)xr[off + 16 + e];
            }
            int off1 = 32 + khi * 8;   // chunk 1: guarded (only k=32..34 valid)
            #pragma unroll
            for (int e = 0; e < 8; e++) {
                int ka = off1 + e, kb2 = off1 + 16 + e;
                A1[e]     = (ka  < D_IN) ? (_Float16)xr[ka]  : (_Float16)0.0f;
                A1[e + 8] = (kb2 < D_IN) ? (_Float16)xr[kb2] : (_Float16)0.0f;
            }
        }
        #pragma unroll
        for (int j = 0; j < 2; j++) {
            v8f acc;
            #pragma unroll
            for (int r = 0; r < 8; r++) acc[r] = biasv[j];
            acc = wmma16(A0, Bf[j][0], acc);
            acc = wmma16(A1, Bf[j][1], acc);
            v8h st;
            #pragma unroll
            for (int r = 0; r < 8; r++) st[r] = (_Float16)acc[r];
            *(v8h*)(xp1 + xp1_base(dir, t, bt, wv * 2 + j, lane)) = st;
        }
    }
}

// ---------------------------------------------------------------------------
// Kernel 2: layer-1 recurrent scan. grid (32 btiles, 2 dirs), 256 threads.
// Per step: gates = xp1[t] + h @ W_hh^T via WMMA; cell update; h -> LDS (f16).
// xp1 fragments prefetched 2 steps ahead.
// ---------------------------------------------------------------------------
__global__ __launch_bounds__(256)
void lstm1_scan(const _Float16* __restrict__ xp1,
                const float* __restrict__ w_hh_f,
                const float* __restrict__ w_hh_b,
                _Float16* __restrict__ out1)
{
    const int dir = blockIdx.y, bt = blockIdx.x;
    const float* whh = dir ? w_hh_b : w_hh_f;   // [256][64]
    __shared__ _Float16 hlds[16 * 72];          // h, row stride 72 (conflict pad)
    __shared__ float    glds[16 * 32 * 8];      // gates in fragment order [nt][lane][r]

    const int tid = threadIdx.x, lane = tid & 31, wv = tid >> 5;
    const int nif = lane & 15, khi = lane >> 4;

    // Resident W_hh B-fragments (B[k][n] = whh[g][k]), K=64 -> 2 chunks
    v16h Bf[2][2];
    #pragma unroll
    for (int j = 0; j < 2; j++) {
        int g = (wv * 2 + j) * 16 + nif;
        #pragma unroll
        for (int kc = 0; kc < 2; kc++) {
            int kb = kc * 32 + khi * 8;
            v16h f;
            #pragma unroll
            for (int e = 0; e < 16; e++) f[e] = (_Float16)whh[g * H1 + kb + e + (e & 8)];
            Bf[j][kc] = f;
        }
    }

    for (int i = tid; i < 16 * 72; i += 256) hlds[i] = (_Float16)0.0f;
    float cst[4] = {0.f, 0.f, 0.f, 0.f};
    const int crow = (tid * 4) >> 6;      // cell row (0..15)
    const int cjb  = (tid * 4) & 63;      // cell hidden base (mult of 4)
    const int b0   = bt * 16;
    __syncthreads();

    auto do_step = [&](int s, v8h& x0, v8h& x1) {
        const int t = dir ? (T_LEN - 1 - s) : s;
        __syncthreads();                              // h (and glds free) ready
        v8f acc0, acc1;
        #pragma unroll
        for (int r = 0; r < 8; r++) { acc0[r] = (float)x0[r]; acc1[r] = (float)x1[r]; }
        const int sp = s + 2;                         // prefetch 2 steps ahead
        if (sp < T_LEN) {
            int tp = dir ? (T_LEN - 1 - sp) : sp;
            x0 = *(const v8h*)(xp1 + xp1_base(dir, tp, bt, wv * 2,     lane));
            x1 = *(const v8h*)(xp1 + xp1_base(dir, tp, bt, wv * 2 + 1, lane));
        }
        // gather A fragments (h) from LDS: two b128 runs per chunk
        v16h A0, A1;
        {
            const _Float16* hp = hlds + nif * 72 + khi * 8;
            v8h lo = *(const v8h*)(hp);
            v8h hi = *(const v8h*)(hp + 16);
            #pragma unroll
            for (int e = 0; e < 8; e++) { A0[e] = lo[e]; A0[e + 8] = hi[e]; }
            lo = *(const v8h*)(hp + 32);
            hi = *(const v8h*)(hp + 48);
            #pragma unroll
            for (int e = 0; e < 8; e++) { A1[e] = lo[e]; A1[e + 8] = hi[e]; }
        }
        acc0 = wmma16(A0, Bf[0][0], acc0);
        acc0 = wmma16(A1, Bf[0][1], acc0);
        acc1 = wmma16(A0, Bf[1][0], acc1);
        acc1 = wmma16(A1, Bf[1][1], acc1);
        *(v8f*)(glds + ((wv * 2    ) * 32 + lane) * 8) = acc0;
        *(v8f*)(glds + ((wv * 2 + 1) * 32 + lane) * 8) = acc1;
        __syncthreads();                              // gates ready
        // cell update: 4 (row, j) elements per thread
        v4h hpk;
        #pragma unroll
        for (int q = 0; q < 4; q++) {
            int j = cjb + q;
            auto gidx = [&](int col) {
                int nt = col >> 4, n = col & 15;
                int ln = n + ((crow & 8) ? 16 : 0);
                return (nt * 32 + ln) * 8 + (crow & 7);
            };
            float gi = glds[gidx(j)];
            float gf = glds[gidx(64 + j)];
            float gg = glds[gidx(128 + j)];
            float go = glds[gidx(192 + j)];
            float cn = sigf(gf) * cst[q] + sigf(gi) * tanh_fast(gg);
            float hn = sigf(go) * tanh_fast(cn);
            cst[q] = cn;
            hlds[crow * 72 + j] = (_Float16)hn;
            hpk[q] = (_Float16)hn;
        }
        *(v4h*)(out1 + (((size_t)(b0 + crow)) * T_LEN + t) * 128 + dir * 64 + cjb) = hpk;
    };

    // double-buffered xp fragments (2 steps deep, no dynamic indexing)
    v8h xa0, xa1, xb0, xb1;
    {
        int t0 = dir ? T_LEN - 1 : 0;
        int t1 = dir ? T_LEN - 2 : 1;
        xa0 = *(const v8h*)(xp1 + xp1_base(dir, t0, bt, wv * 2,     lane));
        xa1 = *(const v8h*)(xp1 + xp1_base(dir, t0, bt, wv * 2 + 1, lane));
        xb0 = *(const v8h*)(xp1 + xp1_base(dir, t1, bt, wv * 2,     lane));
        xb1 = *(const v8h*)(xp1 + xp1_base(dir, t1, bt, wv * 2 + 1, lane));
    }
    for (int s = 0; s < T_LEN; s += 2) {
        do_step(s,     xa0, xa1);
        do_step(s + 1, xb0, xb1);
    }
}

// ---------------------------------------------------------------------------
// Kernel 3: layer-2 input projection. xp2[dir][t][b][12] = out1 @ W2_ih^T + biases.
// grid (2048, 2 dirs), 256 threads; each wave = one 16-row M-tile (fixed b, 16 t).
// ---------------------------------------------------------------------------
__global__ __launch_bounds__(256)
void lstm2_proj(const _Float16* __restrict__ out1,
                const float* __restrict__ w_ih_f,
                const float* __restrict__ b_ih_f, const float* __restrict__ b_hh_f,
                const float* __restrict__ w_ih_b,
                const float* __restrict__ b_ih_b, const float* __restrict__ b_hh_b,
                float* __restrict__ xp2)
{
    const int dir = blockIdx.y;
    const float* w  = dir ? w_ih_b : w_ih_f;
    const float* bi = dir ? b_ih_b : b_ih_f;
    const float* bh = dir ? b_hh_b : b_hh_f;
    const int lane = threadIdx.x & 31, wv = threadIdx.x >> 5;
    const int nif = lane & 15, khi = lane >> 4;
    const int mtile = blockIdx.x * 8 + wv;          // 0..16383
    const int m0 = mtile * 16;

    v16h Bf[4];
    float biasv = (nif < G2) ? (bi[nif] + bh[nif]) : 0.0f;
    #pragma unroll
    for (int kc = 0; kc < 4; kc++) {
        int kb = kc * 32 + khi * 8;
        v16h f;
        #pragma unroll
        for (int e = 0; e < 16; e++)
            f[e] = (nif < G2) ? (_Float16)w[nif * K2 + kb + e + (e & 8)] : (_Float16)0.0f;
        Bf[kc] = f;
    }

    const _Float16* arow = out1 + (size_t)(m0 + nif) * K2;
    v8f acc;
    #pragma unroll
    for (int r = 0; r < 8; r++) acc[r] = biasv;
    #pragma unroll
    for (int kc = 0; kc < 4; kc++) {
        int off = kc * 32 + khi * 8;
        v8h lo = *(const v8h*)(arow + off);
        v8h hi = *(const v8h*)(arow + off + 16);
        v16h A;
        #pragma unroll
        for (int e = 0; e < 8; e++) { A[e] = lo[e]; A[e + 8] = hi[e]; }
        acc = wmma16(A, Bf[kc], acc);
    }
    if (nif < G2) {
        #pragma unroll
        for (int r = 0; r < 8; r++) {
            int mm = m0 + r + khi * 8;
            int b = mm >> 9, t = mm & 511;          // m = b*T + t, T=512
            xp2[(((size_t)dir * T_LEN + t) * B_SZ + b) * 12 + nif] = acc[r];
        }
    }
}

// ---------------------------------------------------------------------------
// Kernel 4: layer-2 recurrent scan (H=3). grid (2 dirs), 512 threads (= rows).
// Pure per-thread scan, state in registers, final output written directly.
// ---------------------------------------------------------------------------
__global__ __launch_bounds__(512)
void lstm2_scan(const float* __restrict__ xp2,
                const float* __restrict__ w_hh_f,
                const float* __restrict__ w_hh_b,
                float* __restrict__ out)
{
    const int dir = blockIdx.x;
    const int b = threadIdx.x;
    const float* whh = dir ? w_hh_b : w_hh_f;   // [12][3]
    float w[12][3];
    #pragma unroll
    for (int n = 0; n < 12; n++)
        #pragma unroll
        for (int j = 0; j < 3; j++) w[n][j] = whh[n * 3 + j];

    float h[3] = {0.f, 0.f, 0.f}, c[3] = {0.f, 0.f, 0.f};
    for (int s = 0; s < T_LEN; s++) {
        int t = dir ? (T_LEN - 1 - s) : s;
        const float* gp = xp2 + (((size_t)dir * T_LEN + t) * B_SZ + b) * 12;
        float4 ga = *(const float4*)(gp);
        float4 gb = *(const float4*)(gp + 4);
        float4 gc = *(const float4*)(gp + 8);
        float g[12] = {ga.x, ga.y, ga.z, ga.w, gb.x, gb.y, gb.z, gb.w,
                       gc.x, gc.y, gc.z, gc.w};
        #pragma unroll
        for (int n = 0; n < 12; n++)
            g[n] += w[n][0] * h[0] + w[n][1] * h[1] + w[n][2] * h[2];
        #pragma unroll
        for (int j = 0; j < 3; j++) {
            float cn = sigf(g[3 + j]) * c[j] + sigf(g[j]) * tanh_fast(g[6 + j]);
            h[j] = sigf(g[9 + j]) * tanh_fast(cn);
            c[j] = cn;
        }
        float* op = out + (((size_t)b * T_LEN + t) * 6 + dir * 3);
        op[0] = h[0]; op[1] = h[1]; op[2] = h[2];
    }
}

extern "C" void kernel_launch(void* const* d_in, const int* in_sizes, int n_in,
                              void* d_out, int out_size, void* d_ws, size_t ws_size,
                              hipStream_t stream) {
    (void)in_sizes; (void)n_in; (void)out_size; (void)ws_size;
    const float* x       = (const float*)d_in[0];
    const float* w1_ih_f = (const float*)d_in[1];
    const float* w1_hh_f = (const float*)d_in[2];
    const float* b1_ih_f = (const float*)d_in[3];
    const float* b1_hh_f = (const float*)d_in[4];
    const float* w1_ih_b = (const float*)d_in[5];
    const float* w1_hh_b = (const float*)d_in[6];
    const float* b1_ih_b = (const float*)d_in[7];
    const float* b1_hh_b = (const float*)d_in[8];
    const float* w2_ih_f = (const float*)d_in[9];
    const float* w2_hh_f = (const float*)d_in[10];
    const float* b2_ih_f = (const float*)d_in[11];
    const float* b2_hh_f = (const float*)d_in[12];
    const float* w2_ih_b = (const float*)d_in[13];
    const float* w2_hh_b = (const float*)d_in[14];
    const float* b2_ih_b = (const float*)d_in[15];
    const float* b2_hh_b = (const float*)d_in[16];

    _Float16* xp1  = (_Float16*)d_ws;                              // 256 MB
    _Float16* out1 = (_Float16*)((char*)d_ws + OUT1_OFF);          //  64 MB
    float*    xp2  = (float*)   ((char*)d_ws + XP2_OFF);           //  24 MB
    float*    out  = (float*)d_out;

    lstm1_proj<<<dim3(32, 32, 2), 256, 0, stream>>>(
        x, w1_ih_f, b1_ih_f, b1_hh_f, w1_ih_b, b1_ih_b, b1_hh_b, xp1);
    lstm1_scan<<<dim3(32, 2), 256, 0, stream>>>(xp1, w1_hh_f, w1_hh_b, out1);
    lstm2_proj<<<dim3(2048, 2), 256, 0, stream>>>(
        out1, w2_ih_f, b2_ih_f, b2_hh_f, w2_ih_b, b2_ih_b, b2_hh_b, xp2);
    lstm2_scan<<<dim3(2), 512, 0, stream>>>(xp2, w2_hh_f, w2_hh_b, out);
}
